// FrameTransformer_18210661335563
// MI455X (gfx1250) — matrix operand order repaired and checked by reading, not compiled, not run on hardware
//
#include <hip/hip_runtime.h>
#include <hip/hip_bf16.h>

// ---------------------------------------------------------------------------
// FrameTransformer decoder block for MI455X (gfx1250).
// All GEMMs run on v_wmma_f32_16x16x32_bf16 (f32->bf16 native cvt while
// staging tiles to LDS, f32 accumulators). wave32, 8 waves per 256-thread
// block, 128x64x32 tiles, 4 WMMAs per wave per K-step.
// ---------------------------------------------------------------------------

#define BB    2
#define CHAN  16
#define BINSD 1024
#define CROPD 1024
#define NB_H  4
#define HD    256
#define FFD   2048
#define ROWS  (BB * CROPD)          // 2048 = b*w rows
#define WC    (1024L * 1024L)       // per-batch [w,c] plane (floats)
#define WWQ   (1024L * 1024L)       // per-(b,n) [w,w] plane (floats)

typedef __bf16 v16bf __attribute__((ext_vector_type(16)));
typedef float  v8f   __attribute__((ext_vector_type(8)));

struct u4 { unsigned int x, y, z, w; };
struct f4 { float x, y, z, w; };

static __device__ __forceinline__ unsigned int pack2_bf16(float lo, float hi) {
  union { __bf16 h[2]; unsigned int u; } p;
  p.h[0] = (__bf16)lo;              // native cvt, RNE
  p.h[1] = (__bf16)hi;
  return p.u;
}

// ---------------------------------------------------------------------------
// Generic WMMA GEMM:  C[m,n] (op)= sum_k A[m,k] * Bt[n,k]  (+bias[n])
// A, Bt are f32 row-major with K contiguous; converted to bf16 into LDS.
// Block tile 128x64, K-step 32. 8 waves in 4(m)x2(n) grid; each wave owns a
// 32x32 output patch: 2 A frags x 2 B frags -> 4 WMMAs per K-step.
// z decomposed as z1 = z / Zinner, z2 = z % Zinner for per-head addressing.
// flags: bit0 = relu, bit1 = square (after relu), bit2 = accumulate into C.
// Requires M % 128 == 0, N % 64 == 0, K % 32 == 0, 16B-aligned A/Bt rows.
// ---------------------------------------------------------------------------
__global__ __launch_bounds__(256) void gemm_bf16_wmma(
    const float* __restrict__ A,  int lda, long As1, long As2,
    const float* __restrict__ Bt, int ldb, long Bs1, long Bs2,
    float* __restrict__ C,        int ldc, long Cs1, long Cs2,
    const float* __restrict__ bias,
    int M, int N, int K, int Zinner, int flags)
{
  __shared__ unsigned int As[128 * 16];  // 128 rows x 16 bf16-pairs (K=32)
  __shared__ unsigned int Bs[64 * 16];   // 64 cols x 16 bf16-pairs

  const int z  = blockIdx.z;
  const int z1 = z / Zinner, z2 = z % Zinner;
  A  += z1 * As1 + z2 * As2;
  Bt += z1 * Bs1 + z2 * Bs2;
  C  += z1 * Cs1 + z2 * Cs2;

  const int mBase = blockIdx.y * 128;
  const int nBase = blockIdx.x * 64;

  const int tid   = threadIdx.x;
  const int lane  = tid & 31;
  const int wave  = tid >> 5;         // 8 waves
  const int waveM = wave >> 1;        // 0..3  (32-row strip)
  const int waveN = wave & 1;         // 0..1  (32-col strip)

  const int r16 = lane & 15;
  const int hi  = lane >> 4;          // lane half

  v8f acc00 = {}, acc01 = {}, acc10 = {}, acc11 = {};

  for (int k0 = 0; k0 < K; k0 += 32) {
    // stage A tile (128x32): each thread converts 2 quads (4 pairs each)
    #pragma unroll
    for (int it = 0; it < 2; ++it) {
      int q = tid + it * 256;               // 512 quads
      int row = q >> 2, kq = q & 3;
      const f4* s = (const f4*)(A + (long)(mBase + row) * lda + k0 + kq * 8);
      f4 a = s[0], b = s[1];
      u4 o; o.x = pack2_bf16(a.x, a.y); o.y = pack2_bf16(a.z, a.w);
            o.z = pack2_bf16(b.x, b.y); o.w = pack2_bf16(b.z, b.w);
      *(u4*)&As[row * 16 + kq * 4] = o;
    }
    // stage Bt tile (64x32): one quad per thread
    {
      int q = tid;                          // 256 quads
      int col = q >> 2, kq = q & 3;
      const f4* s = (const f4*)(Bt + (long)(nBase + col) * ldb + k0 + kq * 8);
      f4 a = s[0], b = s[1];
      u4 o; o.x = pack2_bf16(a.x, a.y); o.y = pack2_bf16(a.z, a.w);
            o.z = pack2_bf16(b.x, b.y); o.w = pack2_bf16(b.z, b.w);
      *(u4*)&Bs[col * 16 + kq * 4] = o;
    }
    __syncthreads();

    // A fragments (16-bit A 16x32, contiguous k order used here, same as B):
    // lanes 0-15: row=lane, v0..7 = K0..15
    // lanes 16-31: same rows,  v0..7 = K16..31
    union { unsigned int u[8]; v16bf v; } af0, af1, bf0, bf1;
    const int arow0 = waveM * 32 + r16;
    const int akb   = hi * 8;               // pair base 0 or 8
    #pragma unroll
    for (int q = 0; q < 4; ++q) {
      af0.u[q]     = As[arow0 * 16 + akb + q];
      af0.u[4 + q] = As[arow0 * 16 + 4 + akb + q];
      af1.u[q]     = As[(arow0 + 16) * 16 + akb + q];
      af1.u[4 + q] = As[(arow0 + 16) * 16 + 4 + akb + q];
    }
    // B fragments: lane = column; lanes 0-15 hold K0..15, lanes 16-31 K16..31
    const int bkb   = hi * 8;
    const int bcol0 = waveN * 32 + r16;
    #pragma unroll
    for (int q = 0; q < 8; ++q) {
      bf0.u[q] = Bs[bcol0 * 16 + bkb + q];
      bf1.u[q] = Bs[(bcol0 + 16) * 16 + bkb + q];
    }

    acc00 = __builtin_amdgcn_wmma_f32_16x16x32_bf16(false, af0.v, false, bf0.v,
                                                    (short)0, acc00, false, false);
    acc01 = __builtin_amdgcn_wmma_f32_16x16x32_bf16(false, af0.v, false, bf1.v,
                                                    (short)0, acc01, false, false);
    acc10 = __builtin_amdgcn_wmma_f32_16x16x32_bf16(false, af1.v, false, bf0.v,
                                                    (short)0, acc10, false, false);
    acc11 = __builtin_amdgcn_wmma_f32_16x16x32_bf16(false, af1.v, false, bf1.v,
                                                    (short)0, acc11, false, false);
    __syncthreads();
  }

  // C/D layout: lane half selects row group (+0 / +8), vgpr r = row offset
  const int col0 = nBase + waveN * 32 + r16;
  const float b0 = bias ? bias[col0]      : 0.f;
  const float b1 = bias ? bias[col0 + 16] : 0.f;
  #pragma unroll
  for (int i = 0; i < 2; ++i) {
    const v8f a0 = i ? acc10 : acc00;
    const v8f a1 = i ? acc11 : acc01;
    const int rowBase = mBase + waveM * 32 + i * 16 + hi * 8;
    #pragma unroll
    for (int r = 0; r < 8; ++r) {
      float v0 = a0[r] + b0;
      float v1 = a1[r] + b1;
      if (flags & 1) { v0 = fmaxf(v0, 0.f); v1 = fmaxf(v1, 0.f); }
      if (flags & 2) { v0 *= v0; v1 *= v1; }
      float* cp = C + (long)(rowBase + r) * ldc;
      if (flags & 4) { cp[col0] += v0; cp[col0 + 16] += v1; }
      else           { cp[col0]  = v0; cp[col0 + 16]  = v1; }
    }
  }
}

// ---------------------------------------------------------------------------
// in/skip projection: out[b,w,f] = sum_c in4[b,c,f,w] * wv[c]
// ---------------------------------------------------------------------------
__global__ __launch_bounds__(256) void project_kernel(
    const float* __restrict__ in4, float* __restrict__ out,
    const float* __restrict__ wv)
{
  int t = blockIdx.x * 256 + threadIdx.x;      // 2M threads
  int w = t & 1023, f = (t >> 10) & 1023, b = t >> 20;
  const float* p = in4 + ((long)b * CHAN) * WC + (long)f * 1024 + w;
  float acc = 0.f;
  #pragma unroll
  for (int c = 0; c < CHAN; ++c) acc += p[(long)c * WC] * wv[c];
  out[(long)b * WC + (long)w * 1024 + f] = acc;
}

// ---------------------------------------------------------------------------
// LayerNorm over last (contiguous) dim D; one block per row.
// ---------------------------------------------------------------------------
__global__ __launch_bounds__(256) void ln_kernel(
    const float* __restrict__ in, float* __restrict__ out,
    const float* __restrict__ g, const float* __restrict__ b, int D)
{
  __shared__ float red[256];
  const float* x = in + (long)blockIdx.x * D;
  float* y = out + (long)blockIdx.x * D;
  float s = 0.f, s2 = 0.f;
  for (int i = threadIdx.x; i < D; i += 256) { float v = x[i]; s += v; s2 += v * v; }
  red[threadIdx.x] = s; __syncthreads();
  for (int o = 128; o > 0; o >>= 1) { if (threadIdx.x < o) red[threadIdx.x] += red[threadIdx.x + o]; __syncthreads(); }
  float mean = red[0] / D; __syncthreads();
  red[threadIdx.x] = s2; __syncthreads();
  for (int o = 128; o > 0; o >>= 1) { if (threadIdx.x < o) red[threadIdx.x] += red[threadIdx.x + o]; __syncthreads(); }
  float var  = red[0] / D - mean * mean;
  float rstd = rsqrtf(var + 1e-5f);
  for (int i = threadIdx.x; i < D; i += 256)
    y[i] = (x[i] - mean) * rstd * g[i] + b[i];
}

// ---------------------------------------------------------------------------
// Depthwise conv along W on [B, W, C] layout (C contiguous): zero pad.
// ---------------------------------------------------------------------------
__global__ __launch_bounds__(256) void dwconv_kernel(
    const float* __restrict__ in, float* __restrict__ out,
    const float* __restrict__ wt, const float* __restrict__ bias,
    int Bn, int W, int C, int KW, int pad)
{
  long t = (long)blockIdx.x * 256 + threadIdx.x;
  long total = (long)Bn * W * C;
  if (t >= total) return;
  int c = (int)(t % C);
  long bw = t / C;
  int w = (int)(bw % W);
  int b = (int)(bw / W);
  const float* base = in + (long)b * W * C + c;
  float acc = bias ? bias[c] : 0.f;
  for (int j = 0; j < KW; ++j) {
    int wp = w + j - pad;
    if (wp >= 0 && wp < W) acc += base[(long)wp * C] * wt[c * KW + j];
  }
  out[t] = acc;
}

// vt[bn, h, kw] = vconv[b, kw, n*256+h]
__global__ __launch_bounds__(256) void transpose_vt_kernel(
    const float* __restrict__ vc, float* __restrict__ vt)
{
  int t = blockIdx.x * 256 + threadIdx.x;      // 2M
  int kw = t & 1023, h = (t >> 10) & 255, bn = t >> 18;
  int b = bn >> 2, n = bn & 3;
  vt[t] = vc[(long)b * WC + (long)kw * 1024 + n * 256 + h];
}

// ert[n, h] = er[h, n]   (er is [256, 1024])
__global__ __launch_bounds__(256) void transpose_er_kernel(
    const float* __restrict__ er, float* __restrict__ ert)
{
  int t = blockIdx.x * 256 + threadIdx.x;      // 262144
  int h = t & 255, n = t >> 8;
  ert[t] = er[h * 1024 + n];
}

// softmax with skew: logits[ki] = (qk[r,ki] + qe[bn,ki,qi]) / 32
__global__ __launch_bounds__(256) void softmax_skew_kernel(
    const float* __restrict__ qk, const float* __restrict__ qe,
    float* __restrict__ a)
{
  __shared__ float red[256];
  int r = blockIdx.x;                  // r = bn*1024 + qi, 8192 rows
  int bn = r >> 10, qi = r & 1023;
  const float* qkr = qk + (long)r * 1024;
  const float* qec = qe + (long)bn * WWQ + qi;   // stride 1024 over ki
  float vals[4];
  float mx = -1e30f;
  #pragma unroll
  for (int i = 0; i < 4; ++i) {
    int ki = threadIdx.x + i * 256;
    float l = (qkr[ki] + qec[(long)ki * 1024]) * 0.03125f;
    vals[i] = l; mx = fmaxf(mx, l);
  }
  red[threadIdx.x] = mx; __syncthreads();
  for (int o = 128; o > 0; o >>= 1) { if (threadIdx.x < o) red[threadIdx.x] = fmaxf(red[threadIdx.x], red[threadIdx.x + o]); __syncthreads(); }
  float M = red[0]; __syncthreads();
  float s = 0.f;
  #pragma unroll
  for (int i = 0; i < 4; ++i) { vals[i] = __expf(vals[i] - M); s += vals[i]; }
  red[threadIdx.x] = s; __syncthreads();
  for (int o = 128; o > 0; o >>= 1) { if (threadIdx.x < o) red[threadIdx.x] += red[threadIdx.x + o]; __syncthreads(); }
  float inv = 1.f / red[0];
  #pragma unroll
  for (int i = 0; i < 4; ++i) {
    int ki = threadIdx.x + i * 256;
    a[(long)r * 1024 + ki] = vals[i] * inv;
  }
}

// out[b,0,f,w] = x[b,w,f]
__global__ __launch_bounds__(256) void final_out_kernel(
    const float* __restrict__ x, float* __restrict__ out)
{
  int t = blockIdx.x * 256 + threadIdx.x;
  int w = t & 1023, f = (t >> 10) & 1023, b = t >> 20;
  out[(long)b * WC + (long)f * 1024 + w] = x[(long)b * WC + (long)w * 1024 + f];
}

// ---------------------------------------------------------------------------
// Host orchestration
// ---------------------------------------------------------------------------
extern "C" void kernel_launch(void* const* d_in, const int* in_sizes, int n_in,
                              void* d_out, int out_size, void* d_ws, size_t ws_size,
                              hipStream_t stream) {
  (void)in_sizes; (void)n_in; (void)out_size; (void)ws_size;
  // Input order (setup_inputs insertion order, recursive):
  // 0:x 1:skip 2:in_w 3:skip_w 4..18:sa1 19..33:ea1 34..48:sa2 49..63:ea2
  // 64:c1L_dw 65:c1L_pw 66:c1R_dw 67:c1R_pw 68:c2_dw 69:c2_pw 70:c3 71:c4
  // 72..83: n1g n1b n2g n2b n3g n3b n4g n4b n5g n5b n6g n6b
  auto P = [&](int i) { return (const float*)d_in[i]; };

  float* ws = (float*)d_ws;
  size_t off = 0;
  auto alloc = [&](size_t n) { float* p = ws + off; off += n; return p; };
  float* xbuf  = alloc(2 * WC);
  float* skipb = alloc(2 * WC);
  float* hbuf  = alloc(2 * WC);
  float* tmp   = alloc(2 * WC);
  float* qc    = alloc(2 * WC);
  float* kc    = alloc(2 * WC);
  float* vc    = alloc(2 * WC);
  float* vt    = alloc(2 * WC);          // [8, 256, 1024]
  float* ert   = alloc(1024 * 256);
  float* qe    = alloc(8 * WWQ);         // reused as hsum [2048,4096] and t
  float* qk    = alloc(8 * WWQ);         // reused as h3   [2048,4096]
  float* abuf  = alloc(8 * WWQ);         // reused as dwconv2 output
  float* obuf  = alloc(2 * WC);
  // total ~44.3M floats (~177 MB) of workspace

  auto gemm = [&](const float* A, int lda, long As1, long As2,
                  const float* Bt, int ldb, long Bs1, long Bs2,
                  float* C, int ldc, long Cs1, long Cs2,
                  const float* bias, int M, int N, int K, int Z, int Zi, int fl) {
    dim3 g(N / 64, M / 128, Z);
    gemm_bf16_wmma<<<g, 256, 0, stream>>>(A, lda, As1, As2, Bt, ldb, Bs1, Bs2,
                                          C, ldc, Cs1, Cs2, bias, M, N, K, Zi, fl);
  };
  auto ln = [&](const float* in, float* out, const float* g, const float* b, int D) {
    ln_kernel<<<ROWS, 256, 0, stream>>>(in, out, g, b, D);
  };
  auto dwconv = [&](const float* in, float* out, const float* wt, const float* bias,
                    int C, int KW) {
    long total = (long)BB * 1024 * C;
    dwconv_kernel<<<(int)((total + 255) / 256), 256, 0, stream>>>(
        in, out, wt, bias, BB, 1024, C, KW, KW / 2);
  };

  // Full attention block; accumulates output projection into xbuf.
  auto attn = [&](int pb, const float* hq, const float* mem) {
    // q/k/v: linear (+bias) then depthwise conv k=3 pad=1
    gemm(hq,  1024, 0, 0, P(pb + 0), 1024, 0, 0, tmp, 1024, 0, 0, P(pb + 1),
         ROWS, 1024, 1024, 1, 1, 0);
    dwconv(tmp, qc, P(pb + 2), P(pb + 3), 1024, 3);
    gemm(mem, 1024, 0, 0, P(pb + 4), 1024, 0, 0, tmp, 1024, 0, 0, P(pb + 5),
         ROWS, 1024, 1024, 1, 1, 0);
    dwconv(tmp, kc, P(pb + 6), P(pb + 7), 1024, 3);
    gemm(mem, 1024, 0, 0, P(pb + 8), 1024, 0, 0, tmp, 1024, 0, 0, P(pb + 9),
         ROWS, 1024, 1024, 1, 1, 0);
    dwconv(tmp, vc, P(pb + 10), P(pb + 11), 1024, 3);

    transpose_vt_kernel<<<8192, 256, 0, stream>>>(vc, vt);
    transpose_er_kernel<<<1024, 256, 0, stream>>>(P(pb + 14), ert);

    // qe[bn, w, k] = q[bn] @ ert^T   (A off = b*WC + n*256)
    gemm(qc, 1024, WC, 256, ert, 256, 0, 0, qe, 1024, 4 * WWQ, WWQ,
         nullptr, 1024, 1024, 256, 8, 4, 0);
    // qk[bn, q, k] = q[bn] @ k[bn]^T
    gemm(qc, 1024, WC, 256, kc, 1024, WC, 256, qk, 1024, 4 * WWQ, WWQ,
         nullptr, 1024, 1024, 256, 8, 4, 0);
    // softmax with skew -> abuf
    softmax_skew_kernel<<<BB * NB_H * 1024, 256, 0, stream>>>(qk, qe, abuf);
    // o[bn, q, h] = a[bn] @ vt[bn]^T ; scattered into obuf [b,w,c]
    gemm(abuf, 1024, 4 * WWQ, WWQ, vt, 1024, 4 * HD * 1024L, HD * 1024L,
         obuf, 1024, WC, 256, nullptr, 1024, HD, 1024, 8, 4, 0);
    // out proj (+bias), residual-accumulate into xbuf
    gemm(obuf, 1024, 0, 0, P(pb + 12), 1024, 0, 0, xbuf, 1024, 0, 0, P(pb + 13),
         ROWS, 1024, 1024, 1, 1, 4);
  };

  // 1. channel projections
  project_kernel<<<8192, 256, 0, stream>>>(P(0), xbuf,  P(2));
  project_kernel<<<8192, 256, 0, stream>>>(P(1), skipb, P(3));

  // 2. attn pair 1
  ln(xbuf, hbuf, P(72), P(73), 1024);
  attn(4,  hbuf, hbuf);    // sa1
  attn(19, hbuf, skipb);   // ea1

  // 3. conv-FFN
  float* hsum = qe;        // [2048, 4096]
  float* h3   = qk;        // [2048, 4096]
  float* dw2  = abuf;      // [2048, 4096]
  ln(xbuf, hbuf, P(74), P(75), 1024);
  dwconv(hbuf, tmp, P(64), nullptr, 1024, 11);
  gemm(tmp, 1024, 0, 0, P(65), 1024, 0, 0, hsum, 4096, 0, 0, nullptr,
       ROWS, 4096, 1024, 1, 1, /*relu*/1);
  dwconv(hbuf, tmp, P(66), nullptr, 1024, 7);
  gemm(tmp, 1024, 0, 0, P(67), 1024, 0, 0, hsum, 4096, 0, 0, nullptr,
       ROWS, 2048, 1024, 1, 1, /*accumulate into first 2048 cols*/4);
  ln(hsum, h3, P(76), P(77), 4096);
  dwconv(h3, dw2, P(68), nullptr, 4096, 7);
  gemm(dw2, 4096, 0, 0, P(69), 4096, 0, 0, xbuf, 1024, 0, 0, nullptr,
       ROWS, 1024, 4096, 1, 1, 4);

  // 4. attn pair 2
  ln(xbuf, hbuf, P(78), P(79), 1024);
  attn(34, hbuf, hbuf);    // sa2
  ln(xbuf, hbuf, P(80), P(81), 1024);
  attn(49, hbuf, skipb);   // ea2

  // 5. squared-ReLU FFN
  float* tbuf = qe;        // [2048, 2048]
  ln(xbuf, hbuf, P(82), P(83), 1024);
  gemm(hbuf, 1024, 0, 0, P(70), 1024, 0, 0, tbuf, 2048, 0, 0, nullptr,
       ROWS, 2048, 1024, 1, 1, /*relu^2*/3);
  gemm(tbuf, 2048, 0, 0, P(71), 2048, 0, 0, xbuf, 1024, 0, 0, nullptr,
       ROWS, 1024, 2048, 1, 1, 4);

  // 6. output transpose [b,w,f] -> [b,1,f,w]
  final_out_kernel<<<8192, 256, 0, stream>>>(xbuf, (float*)d_out);
}
